// GRU_82386062672050
// MI455X (gfx1250) — compile-verified, matrix operand order checked by reading, not run on hardware
//
#include <hip/hip_runtime.h>
#include <hip/hip_bf16.h>
#include <math.h>

// ---------------- problem constants ----------------
#define SEQ  2048
#define BAT  64
#define DIN  512
#define HID  512
#define NWG  32          // persistent workgroups for recurrence (each owns 16 h-columns)

typedef __attribute__((ext_vector_type(8)))  __bf16 v8bf;
typedef __attribute__((ext_vector_type(16))) __bf16 v16bf;
typedef __attribute__((ext_vector_type(8)))  float  v8f;
typedef __attribute__((ext_vector_type(4)))  unsigned int u32x4;
typedef __attribute__((ext_vector_type(4)))  int i32x4;
typedef __attribute__((ext_vector_type(8)))  int i32x8;

// Build a 16x32 (A) or 32x16 (B) bf16 WMMA fragment from a row-major row in LDS.
// Lane l (half = l>>4) holds the two contiguous 8-element K runs at half*8 and half*8+16.
__device__ __forceinline__ v16bf make_frag(const __bf16* rowbase, int half) {
  v8bf lo = *(const v8bf*)(rowbase + half * 8);
  v8bf hi = *(const v8bf*)(rowbase + half * 8 + 16);
  return __builtin_shufflevector(lo, hi, 0,1,2,3,4,5,6,7,8,9,10,11,12,13,14,15);
}

__device__ __forceinline__ v8f wmma_bf16(v16bf a, v16bf b, v8f c) {
  return __builtin_amdgcn_wmma_f32_16x16x32_bf16(false, a, false, b, (short)0, c, false, false);
}

__device__ __forceinline__ float sigmoidf_fast(float x) {
  return 1.0f / (1.0f + __expf(-x));
}

// CDNA5 async copy: global -> LDS, 16B per lane, tracked by ASYNCcnt (no VGPR data path).
__device__ __forceinline__ void async_copy_b128(unsigned ldsoff, const void* sbase, unsigned goff) {
  asm volatile("global_load_async_to_lds_b128 %0, %1, %2"
               :: "v"(ldsoff), "v"(goff), "s"(sbase) : "memory");
}
__device__ __forceinline__ unsigned lds_off(const void* p) {
  // generic LDS address = {shared-aperture, 32-bit LDS offset}; low 32 bits are the offset
  return (unsigned)(unsigned long long)p;
}

// ---------------- phase 0a: weight conversion + state init (re-run every launch) --------
__global__ __launch_bounds__(256) void gru_prep(
    const float* __restrict__ W_ir, const float* __restrict__ W_iz, const float* __restrict__ W_in,
    const float* __restrict__ W_hr, const float* __restrict__ W_hz, const float* __restrict__ W_hn,
    __bf16* __restrict__ wi, __bf16* __restrict__ wh,
    float* __restrict__ hping, __bf16* __restrict__ hbf, unsigned* __restrict__ counter)
{
  const size_t n = (size_t)HID * DIN;            // 262144
  size_t i = (size_t)blockIdx.x * blockDim.x + threadIdx.x;
  if (i < n) {
    wi[i]       = (__bf16)W_ir[i];
    wi[n + i]   = (__bf16)W_iz[i];
    wi[2*n + i] = (__bf16)W_in[i];
    wh[i]       = (__bf16)W_hr[i];
    wh[n + i]   = (__bf16)W_hz[i];
    wh[2*n + i] = (__bf16)W_hn[i];
  }
  if (i < (size_t)2 * BAT * HID) {               // h0 = 0, both ping buffers
    hping[i] = 0.0f;
    hbf[i]   = (__bf16)0.0f;
  }
  if (i == 0) *counter = 0u;                     // reset grid barrier every launch
}

// ---------------- phase 0b: x f32 [B][S][D] -> bf16 in GEMM row order [s*B+b][D] ---------
__global__ __launch_bounds__(256) void gru_xcast(
    const float* __restrict__ x, __bf16* __restrict__ xbf)
{
  size_t i = ((size_t)blockIdx.x * 256 + threadIdx.x) * 8;   // flat element index into x
  size_t row = i >> 9;                   // (b*SEQ + s)
  int d = (int)(i & 511);
  int b = (int)(row / SEQ);
  int s = (int)(row % SEQ);
  float4 f0 = *(const float4*)(x + i);
  float4 f1 = *(const float4*)(x + i + 4);
  v8bf o;
  o[0]=(__bf16)f0.x; o[1]=(__bf16)f0.y; o[2]=(__bf16)f0.z; o[3]=(__bf16)f0.w;
  o[4]=(__bf16)f1.x; o[5]=(__bf16)f1.y; o[6]=(__bf16)f1.z; o[7]=(__bf16)f1.w;
  *(v8bf*)(xbf + (((size_t)s * BAT + b) << 9) + d) = o;
}

// ---------------- phase 1: input projections  gbuf[g][s][b][h] = x @ W_i^T + b ----------
// grid: (HID/64, (SEQ*BAT)/64, 3), block 256 (8 waves). 64x64 tile, K-tile 64,
// double-buffered LDS fed by CDNA5 async global->LDS copies (ASYNCcnt pipeline).
__global__ __launch_bounds__(256) void gru_input_gemm(
    const __bf16* __restrict__ xbf,               // [SEQ*BAT][DIN] bf16
    const __bf16* __restrict__ wi,                // [3][HID][DIN] bf16 row-major
    const float* __restrict__ b_ir, const float* __restrict__ b_iz, const float* __restrict__ b_in,
    float* __restrict__ gbuf)                     // [3][SEQ][BAT][HID]
{
  __shared__ __bf16 As[2][64][64];                // 16 KB double-buffered A
  __shared__ __bf16 Bs[2][64][64];                // 16 KB double-buffered B
  const int g  = blockIdx.z;
  const int mb = blockIdx.y;
  const int nb = blockIdx.x;
  const int tid  = threadIdx.x;
  const int lane = tid & 31;
  const int wave = tid >> 5;
  const int wm   = wave & 3;                      // m-tile
  const int wnb  = (wave >> 2) * 32;              // n offset (2 tiles of 16)
  const int half = lane >> 4;
  const int l16  = lane & 15;

  const __bf16* w = wi + (size_t)g * HID * DIN;
  const unsigned aBase = lds_off(&As[0][0][0]);
  const unsigned bBase = lds_off(&Bs[0][0][0]);

  // this thread's two 16B chunks of each 64x64 (8 KB) tile
  const int c0 = tid * 2, c1 = tid * 2 + 1;
  const int r0 = c0 >> 3, s0 = c0 & 7;
  const int r1 = c1 >> 3, s1 = c1 & 7;

  v8f acc0 = {}; v8f acc1 = {};

  auto issue = [&](int st) {
    const int kt = st * 64;
    const unsigned ab = aBase + (unsigned)(st & 1) * 8192u;
    const unsigned bb = bBase + (unsigned)(st & 1) * 8192u;
    async_copy_b128(ab + r0*128 + s0*16, xbf, (unsigned)(((mb*64 + r0)*512 + kt)*2) + s0*16);
    async_copy_b128(ab + r1*128 + s1*16, xbf, (unsigned)(((mb*64 + r1)*512 + kt)*2) + s1*16);
    async_copy_b128(bb + r0*128 + s0*16, w,   (unsigned)(((nb*64 + r0)*512 + kt)*2) + s0*16);
    async_copy_b128(bb + r1*128 + s1*16, w,   (unsigned)(((nb*64 + r1)*512 + kt)*2) + s1*16);
  };

  issue(0);
  for (int st = 0; st < 8; ++st) {                // 8 K-stages of 64
    const int cur = st & 1;
    if (st + 1 < 8) {
      issue(st + 1);
      asm volatile("s_wait_asynccnt 0x4" ::: "memory");  // stage st's 4 copies done
    } else {
      asm volatile("s_wait_asynccnt 0x0" ::: "memory");
    }
    __syncthreads();                              // everyone's copies for stage st landed
#pragma unroll
    for (int ks = 0; ks < 64; ks += 32) {
      v16bf a  = make_frag(&As[cur][wm * 16 + l16][ks], half);
      v16bf b0 = make_frag(&Bs[cur][wnb      + l16][ks], half);
      v16bf b1 = make_frag(&Bs[cur][wnb + 16 + l16][ks], half);
      acc0 = wmma_bf16(a, b0, acc0);
      acc1 = wmma_bf16(a, b1, acc1);
    }
    __syncthreads();                              // done reading before next overwrite
  }

  const float* bi = (g == 0) ? b_ir : (g == 1) ? b_iz : b_in;
  const int col0 = nb * 64 + wnb + l16;
  const int col1 = col0 + 16;
  const float bv0 = bi[col0], bv1 = bi[col1];
  float* outg = gbuf + (size_t)g * SEQ * BAT * HID;
#pragma unroll
  for (int p = 0; p < 8; ++p) {
    int mr = mb * 64 + wm * 16 + p + half * 8;    // flat [s][b] row == m
    float* op = outg + (size_t)mr * HID;
    op[col0] = acc0[p] + bv0;
    op[col1] = acc1[p] + bv1;
  }
}

// ---------------- phase 2: persistent recurrence -----------------------------------------
// NWG workgroups x 128 threads. WG `wg` owns h-columns [wg*16, wg*16+16). W_h slices are
// pinned in LDS; each step one Tensor-Data-Mover op pulls the bf16 h_t tile (64x512) into
// LDS, then 48 WMMAs/wave, gate math, and one agent-scope grid barrier.
__global__ __launch_bounds__(128) void gru_recurrence(
    const float* __restrict__ gbuf,               // [3][SEQ][BAT][HID]
    const __bf16* __restrict__ wh,                // [3][HID][HID]
    const float* __restrict__ b_hr, const float* __restrict__ b_hz, const float* __restrict__ b_hn,
    float* __restrict__ hping,                    // [2][BAT*HID] f32 state
    __bf16* __restrict__ hbf,                     // [2][BAT*HID] bf16 state (TDM source)
    unsigned* __restrict__ counter,
    float* __restrict__ out)                      // [SEQ*BAT*HID] output ++ [BAT*HID] h_last
{
  __shared__ __bf16 hA[BAT][HID];                 // 64 KB: bf16 h_t tile (TDM destination)
  __shared__ __bf16 wS[3][16][HID];               // 48 KB: this WG's W_h row slices
  const int wg   = blockIdx.x;
  const int tid  = threadIdx.x;
  const int lane = tid & 31;
  const int wave = tid >> 5;                      // m-tile 0..3
  const int half = lane >> 4;
  const int l16  = lane & 15;
  const int c    = wg * 16 + l16;                 // this lane's global h-column in C/D

  for (int i = tid * 8; i < 3 * 16 * HID; i += 128 * 8) {
    int g = i / (16 * HID);
    int r = (i / HID) % 16;
    int k = i % HID;
    *(v8bf*)&wS[g][r][k] = *(const v8bf*)(wh + ((size_t)g * HID + wg * 16 + r) * HID + k);
  }
  const float bhr = b_hr[c], bhz = b_hz[c], bhn = b_hn[c];
  __syncthreads();

  const size_t GATE = (size_t)SEQ * BAT * HID;
  const unsigned hAoff = lds_off(&hA[0][0]);

  for (int s = 0; s < SEQ; ++s) {
    const float*  hcur = hping + (size_t)(s & 1) * (BAT * HID);
    float*        hnxt = hping + (size_t)((s + 1) & 1) * (BAT * HID);
    __bf16*       hbfn = hbf   + (size_t)((s + 1) & 1) * (BAT * HID);

    // --- TDM: one wave DMAs the full 64x512 bf16 h_t tile into LDS ---
#if __has_builtin(__builtin_amdgcn_tensor_load_to_lds)
    if (wave == 0) {
      unsigned long long ga =
          (unsigned long long)(hbf + (size_t)(s & 1) * (BAT * HID));
      u32x4 g0;
      g0[0] = 1u;                                  // count=1, user descriptor
      g0[1] = hAoff;                               // lds_addr
      g0[2] = (unsigned)(ga & 0xffffffffu);        // global_addr[31:0]
      g0[3] = (unsigned)((ga >> 32) & 0x01ffffffu) // global_addr[56:32]
              | 0x80000000u;                       // type=2 ("image")
      i32x8 g1;
      g1[0] = 0x00010000;                          // data_size=1 (2 bytes/elem)
      g1[1] = (int)(512u << 16);                   // tensor_dim0 = 512
      g1[2] = (int)(64u  << 16);                   // tensor_dim1 = 64
      g1[3] = (int)(512u << 16);                   // tile_dim0   = 512
      g1[4] = 64;                                  // tile_dim1   = 64 (tile_dim2=0)
      g1[5] = 512;                                 // tensor_dim0_stride = 512
      g1[6] = 0; g1[7] = 0;
      i32x4 z4 = {0, 0, 0, 0};
#if defined(__clang_major__) && __clang_major__ >= 23
      i32x8 z8 = {0, 0, 0, 0, 0, 0, 0, 0};
      __builtin_amdgcn_tensor_load_to_lds(g0, g1, z4, z4, z8, 0);
#else
      __builtin_amdgcn_tensor_load_to_lds(g0, g1, z4, z4, 0);
#endif
      __builtin_amdgcn_s_wait_tensorcnt(0);
    }
#else
    for (int i = tid * 8; i < BAT * HID; i += 128 * 8)
      *(v8bf*)(&hA[0][0] + i) = *(const v8bf*)(hbf + (size_t)(s & 1) * (BAT * HID) + i);
#endif
    // warm L2 with next step's gate rows for this lane's column
    if (s + 1 < SEQ) {
      size_t nidx = ((size_t)(s + 1) * BAT + wave * 16) * HID + c;
      __builtin_prefetch(gbuf + nidx,            0, 0);
      __builtin_prefetch(gbuf + GATE + nidx,     0, 0);
      __builtin_prefetch(gbuf + 2 * GATE + nidx, 0, 0);
    }
    __syncthreads();                               // h_t tile visible to all waves

    // h_t @ W_h{r,z,n}^T for this WG's 16 columns; M-tile = wave
    v8f accR = {}, accZ = {}, accN = {};
#pragma unroll 4
    for (int kt = 0; kt < HID; kt += 32) {
      v16bf a  = make_frag(&hA[wave * 16 + l16][kt], half);
      v16bf br = make_frag(&wS[0][l16][kt], half);
      v16bf bz = make_frag(&wS[1][l16][kt], half);
      v16bf bn = make_frag(&wS[2][l16][kt], half);
      accR = wmma_bf16(a, br, accR);
      accZ = wmma_bf16(a, bz, accZ);
      accN = wmma_bf16(a, bn, accN);
    }

    // gates + state update; C/D layout: VGPR p -> row m = wave*16 + p + 8*half, col = c
#pragma unroll
    for (int p = 0; p < 8; ++p) {
      int m = wave * 16 + p + half * 8;            // batch index
      size_t gidx = ((size_t)s * BAT + m) * HID + c;
      float xr = gbuf[gidx];
      float xz = gbuf[GATE + gidx];
      float xn = gbuf[2 * GATE + gidx];
      float hprev = hcur[m * HID + c];
      float r = sigmoidf_fast(xr + accR[p] + bhr);
      float z = sigmoidf_fast(xz + accZ[p] + bhz);
      float n = tanhf(xn + r * (accN[p] + bhn));
      float hnew = (1.0f - z) * n + z * hprev;
      hnxt[m * HID + c] = hnew;                    // f32 state (exact blend next step)
      hbfn[m * HID + c] = (__bf16)hnew;            // bf16 state (next step's TDM source)
      out[gidx] = hnew;                            // output[s][b][h]
      if (s == SEQ - 1) out[GATE + (size_t)m * HID + c] = hnew;   // h_last
    }

    // grid-wide barrier: release our h_{t+1} stores, acquire everyone else's
    __threadfence();
    __syncthreads();
    if (tid == 0) {
      __hip_atomic_fetch_add(counter, 1u, __ATOMIC_RELEASE, __HIP_MEMORY_SCOPE_AGENT);
      const unsigned target = (unsigned)(s + 1) * NWG;
      while (__hip_atomic_load(counter, __ATOMIC_ACQUIRE, __HIP_MEMORY_SCOPE_AGENT) < target)
        __builtin_amdgcn_s_sleep(1);
    }
    __syncthreads();
  }
}

// ---------------- host-side launcher ------------------------------------------------------
extern "C" void kernel_launch(void* const* d_in, const int* in_sizes, int n_in,
                              void* d_out, int out_size, void* d_ws, size_t ws_size,
                              hipStream_t stream) {
  const float* x    = (const float*)d_in[0];
  const float* W_ir = (const float*)d_in[1];  const float* b_ir = (const float*)d_in[2];
  const float* W_hr = (const float*)d_in[3];  const float* b_hr = (const float*)d_in[4];
  const float* W_iz = (const float*)d_in[5];  const float* b_iz = (const float*)d_in[6];
  const float* W_hz = (const float*)d_in[7];  const float* b_hz = (const float*)d_in[8];
  const float* W_in = (const float*)d_in[9];  const float* b_in = (const float*)d_in[10];
  const float* W_hn = (const float*)d_in[11]; const float* b_hn = (const float*)d_in[12];

  char* ws = (char*)d_ws;
  size_t off = 0;
  unsigned* counter = (unsigned*)(ws + off); off += 256;
  float*    hping   = (float*)  (ws + off); off += 2ull * BAT * HID * 4;   // 256 KB
  __bf16*   hbf     = (__bf16*) (ws + off); off += 2ull * BAT * HID * 2;   // 128 KB
  __bf16*   wi      = (__bf16*) (ws + off); off += 3ull * HID * DIN * 2;   // 1.5 MB
  __bf16*   wh      = (__bf16*) (ws + off); off += 3ull * HID * HID * 2;   // 1.5 MB
  __bf16*   xbf     = (__bf16*) (ws + off); off += (size_t)SEQ * BAT * DIN * 2; // 128 MB
  float*    gbuf    = (float*)  (ws + off);                                // 768 MB

  float* out = (float*)d_out;

  gru_prep<<<(HID * DIN + 255) / 256, 256, 0, stream>>>(
      W_ir, W_iz, W_in, W_hr, W_hz, W_hn, wi, wh, hping, hbf, counter);

  gru_xcast<<<(SEQ * BAT * DIN / 8 + 255) / 256, 256, 0, stream>>>(x, xbf);

  gru_input_gemm<<<dim3(HID / 64, (SEQ * BAT) / 64, 3), 256, 0, stream>>>(
      xbf, wi, b_ir, b_iz, b_in, gbuf);

  gru_recurrence<<<NWG, 128, 0, stream>>>(
      gbuf, wh, b_hr, b_hz, b_hn, hping, hbf, counter, out);
}
